// BiLSTM_CRF_13348758356286
// MI455X (gfx1250) — compile-verified, matrix operand order checked by reading, not run on hardware
//
#include <hip/hip_runtime.h>

// ---------------- types ----------------
typedef __attribute__((ext_vector_type(16))) __bf16 v16bf;
typedef __attribute__((ext_vector_type(8)))  float  v8f;
typedef __attribute__((ext_vector_type(4)))  unsigned int u32x4;

union FragBF { u32x4 u[2]; v16bf v; };
union PackBF { __bf16 h[8]; u32x4 u; };

#define Sd 512
#define Bt 64
#define Ed 128
#define Hd 256
#define G4 1024   // 4*H
#define Td 9

__device__ __forceinline__ float sigf(float x) { return 1.f / (1.f + __expf(-x)); }

// ---------------- kernel 1: embedding gather -> bf16 ----------------
__global__ void embed_kernel(const int* __restrict__ sent,
                             const float* __restrict__ emb,
                             __bf16* __restrict__ x) {
  int i  = blockIdx.x * 256 + threadIdx.x;   // exact grid: S*B*E
  int e  = i & (Ed - 1);
  int sb = i >> 7;                           // E=128
  int tok = sent[sb];
  float v = (tok == 0) ? 0.f : emb[(size_t)tok * Ed + e];   // PAD row = 0
  x[i] = (__bf16)v;
}

// ---------------- kernel 2: f32 -> bf16 weight casts ----------------
__global__ void f2bf_kernel(const float* __restrict__ s, __bf16* __restrict__ d, int n) {
  int i = blockIdx.x * 256 + threadIdx.x;
  if (i < n) d[i] = (__bf16)s[i];
}

// pad W_out (9 x 512) to (16 x 512) bf16, zero rows t>=9
__global__ void woutpad_kernel(const float* __restrict__ w, __bf16* __restrict__ d) {
  int i = blockIdx.x * 256 + threadIdx.x;    // exact grid: 16*512
  int t = i >> 9, k = i & 511;
  d[i] = (__bf16)((t < Td) ? w[t * 512 + k] : 0.f);
}

// ---------------- kernel 3a: full-chip precompute of x @ W_ih^T -----------------
// Output stored per-tile in WMMA C-fragment layout: 16B per lane, 512B per tile,
// tile index (((s*32 + w)*8 + (g*2+n))  with w = nb*4 + mt  (matches recurrence kernel).
__global__ __launch_bounds__(256)
void xgates_kernel(const __bf16* __restrict__ xbf,
                   const __bf16* wih_f, const __bf16* wih_b,
                   __bf16* __restrict__ gout_f, __bf16* __restrict__ gout_b) {
  const int rt  = blockIdx.x;                // row tile over S*B (2048)
  const int dir = blockIdx.y;
  const __bf16* wih = dir ? wih_b : wih_f;
  __bf16* gout = dir ? gout_b : gout_f;

  const int tid = threadIdx.x, lane = tid & 31, wv = tid >> 5;   // 8 waves
  const int m = lane & 15, kh = lane >> 4;
  const int s = rt >> 2, mt = rt & 3;

  const u32x4* pa = (const u32x4*)(xbf + ((size_t)rt * 16 + m) * Ed + kh * 8);
  const u32x4* pb[8];
  #pragma unroll
  for (int i = 0; i < 8; ++i) {
    const int ct = wv * 8 + i;                               // col tile 0..63 of 4H
    pb[i] = (const u32x4*)(wih + (ct * 16 + m) * Ed + kh * 8);
  }

  v8f acc[8];
  #pragma unroll
  for (int i = 0; i < 8; ++i) acc[i] = (v8f){0.f,0.f,0.f,0.f,0.f,0.f,0.f,0.f};

  #pragma unroll
  for (int kc = 0; kc < 4; ++kc) {           // K = E = 128
    FragBF a;
    a.u[0] = pa[kc * 4];
    a.u[1] = pa[kc * 4 + 2];
    #pragma unroll
    for (int i = 0; i < 8; ++i) {
      FragBF b;
      b.u[0] = pb[i][kc * 4];
      b.u[1] = pb[i][kc * 4 + 2];
      acc[i] = __builtin_amdgcn_wmma_f32_16x16x32_bf16(
          false, a.v, false, b.v, (short)0, acc[i], false, false);
    }
  }

  #pragma unroll
  for (int i = 0; i < 8; ++i) {
    const int ct = wv * 8 + i;
    const int g = ct >> 4, colT = ct & 15, nb = colT >> 1, n = colT & 1;
    const int w = nb * 4 + mt;
    PackBF pk;
    #pragma unroll
    for (int j = 0; j < 8; ++j) pk.h[j] = (__bf16)acc[i][j];
    ((u32x4*)gout)[(((size_t)s * 32 + w) * 8 + (g * 2 + n)) * 32 + lane] = pk.u;
  }
}

// ---------------- kernel 3b: recurrence only (h @ W_hh^T), persistent ----------------
__global__ __launch_bounds__(1024)
void bilstm_split_kernel(const __bf16* gates_f, const __bf16* gates_b,
                         const __bf16* whh_f, const float* bias_f,
                         const __bf16* whh_b, const float* bias_b,
                         __bf16* __restrict__ hf, __bf16* __restrict__ hb) {
  __shared__ __bf16 sh_h[Bt * Hd];   // 32 KB

  const int dir = blockIdx.x;
  const __bf16* whh  = dir ? whh_b  : whh_f;
  const float*  bias = dir ? bias_b : bias_f;
  const __bf16* gbuf = dir ? gates_b : gates_f;
  __bf16*       hout = dir ? hb : hf;

  const int tid = threadIdx.x, lane = tid & 31, w = tid >> 5;
  const int mt = w & 3, nb = w >> 2;
  const int m = lane & 15, kh = lane >> 4;

  const int arow = mt * 16 + m;
  const u32x4* ph = (const u32x4*)(sh_h + arow * Hd + kh * 8);

  const u32x4* phh[4][2];
  const u32x4* pg[4][2];
  float bgate[4][2];
  #pragma unroll
  for (int g = 0; g < 4; ++g) {
    #pragma unroll
    for (int n = 0; n < 2; ++n) {
      const int colH = (nb * 2 + n) * 16 + m;
      phh[g][n] = (const u32x4*)(whh + (g * Hd + colH) * Hd + kh * 8);
      pg[g][n]  = (const u32x4*)gbuf + ((size_t)w * 8 + (g * 2 + n)) * 32 + lane;
      bgate[g][n] = bias[g * Hd + colH];
    }
  }
  const int drow0 = mt * 16 + kh * 8;
  __bf16* shh_o0 = sh_h + drow0 * Hd + (nb * 2 + 0) * 16 + m;
  __bf16* shh_o1 = sh_h + drow0 * Hd + (nb * 2 + 1) * 16 + m;
  __bf16* gout0  = hout + (size_t)drow0 * Hd + (nb * 2 + 0) * 16 + m;
  __bf16* gout1  = hout + (size_t)drow0 * Hd + (nb * 2 + 1) * 16 + m;

  for (int i = tid; i < Bt * Hd; i += 1024) sh_h[i] = (__bf16)0.f;   // h0 = 0

  float c[2][8];
  #pragma unroll
  for (int n = 0; n < 2; ++n)
    #pragma unroll
    for (int j = 0; j < 8; ++j) c[n][j] = 0.f;

  for (int t = 0; t < Sd; ++t) {
    const int s = dir ? (Sd - 1 - t) : t;
    const size_t gofs = (size_t)s * 8192;      // step stride in u32x4 units (32*8*32)

    // prefetch next step's gate fragments
    if (t + 1 < Sd) {
      const size_t gnext = (size_t)(dir ? (Sd - 2 - t) : (t + 1)) * 8192;
      #pragma unroll
      for (int g = 0; g < 4; ++g)
        #pragma unroll
        for (int n = 0; n < 2; ++n)
          __builtin_prefetch(pg[g][n] + gnext, 0, 3);
    }
    __syncthreads();                 // publishes previous step's sh_h
    asm volatile("" ::: "memory");   // keep per-step loads in-loop (no LICM/spill)

    // accumulator init = precomputed x-gate fragment
    v8f acc[4][2];
    #pragma unroll
    for (int g = 0; g < 4; ++g) {
      #pragma unroll
      for (int n = 0; n < 2; ++n) {
        PackBF pk;
        pk.u = pg[g][n][gofs];
        #pragma unroll
        for (int j = 0; j < 8; ++j) acc[g][n][j] = (float)pk.h[j];
      }
    }

    // h contribution: K = H = 8 chunks of 32
    #pragma unroll
    for (int kc = 0; kc < 8; ++kc) {
      FragBF a;
      a.u[0] = ph[kc * 4];
      a.u[1] = ph[kc * 4 + 2];
      #pragma unroll
      for (int g = 0; g < 4; ++g) {
        #pragma unroll
        for (int n = 0; n < 2; ++n) {
          FragBF bfr;
          bfr.u[0] = phh[g][n][kc * 4];
          bfr.u[1] = phh[g][n][kc * 4 + 2];
          acc[g][n] = __builtin_amdgcn_wmma_f32_16x16x32_bf16(
              false, a.v, false, bfr.v, (short)0, acc[g][n], false, false);
        }
      }
    }
    __syncthreads();                 // all reads of sh_h complete

    const size_t so = (size_t)s * Bt * Hd;
    #pragma unroll
    for (int n = 0; n < 2; ++n) {
      __bf16* shh_o = n ? shh_o1 : shh_o0;
      __bf16* go_   = n ? gout1  : gout0;
      #pragma unroll
      for (int j = 0; j < 8; ++j) {
        const float gi = sigf (acc[0][n][j] + bgate[0][n]);
        const float gf = sigf (acc[1][n][j] + bgate[1][n]);
        const float gg = tanhf(acc[2][n][j] + bgate[2][n]);
        const float go = sigf (acc[3][n][j] + bgate[3][n]);
        const float cv = gf * c[n][j] + gi * gg;
        c[n][j] = cv;
        const float hv = go * tanhf(cv);
        shh_o[j * Hd]    = (__bf16)hv;
        go_[so + j * Hd] = (__bf16)hv;
      }
    }
  }
}

// ---------------- kernel 3c: fused fallback (small-ws path) ----------------
__global__ __launch_bounds__(1024)
void bilstm_kernel(const __bf16* __restrict__ xbf,
                   const __bf16* wih_f, const __bf16* whh_f, const float* bias_f,
                   const __bf16* wih_b, const __bf16* whh_b, const float* bias_b,
                   __bf16* __restrict__ hf, __bf16* __restrict__ hb) {
  __shared__ __bf16 sh_h[Bt * Hd];
  __shared__ __bf16 sh_x[Bt * Ed];

  const int dir = blockIdx.x;
  const __bf16* wih  = dir ? wih_b  : wih_f;
  const __bf16* whh  = dir ? whh_b  : whh_f;
  const float*  bias = dir ? bias_b : bias_f;
  __bf16*       hout = dir ? hb     : hf;

  const int tid = threadIdx.x, lane = tid & 31, w = tid >> 5;
  const int mt = w & 3, nb = w >> 2;
  const int m = lane & 15, kh = lane >> 4;

  const int arow = mt * 16 + m;
  const u32x4* px = (const u32x4*)(sh_x + arow * Ed + kh * 8);
  const u32x4* ph = (const u32x4*)(sh_h + arow * Hd + kh * 8);

  const u32x4* pih[4][2];
  const u32x4* phh[4][2];
  float bgate[4][2];
  #pragma unroll
  for (int g = 0; g < 4; ++g) {
    #pragma unroll
    for (int n = 0; n < 2; ++n) {
      const int colH = (nb * 2 + n) * 16 + m;
      const int wrow = g * Hd + colH;
      pih[g][n] = (const u32x4*)(wih + wrow * Ed + kh * 8);
      phh[g][n] = (const u32x4*)(whh + wrow * Hd + kh * 8);
      bgate[g][n] = bias[g * Hd + colH];
    }
  }
  const int drow0 = mt * 16 + kh * 8;
  __bf16* shh_o0 = sh_h + drow0 * Hd + (nb * 2 + 0) * 16 + m;
  __bf16* shh_o1 = sh_h + drow0 * Hd + (nb * 2 + 1) * 16 + m;
  __bf16* gout0  = hout + (size_t)drow0 * Hd + (nb * 2 + 0) * 16 + m;
  __bf16* gout1  = hout + (size_t)drow0 * Hd + (nb * 2 + 1) * 16 + m;

  for (int i = tid; i < Bt * Hd; i += 1024) sh_h[i] = (__bf16)0.f;

  float c[2][8];
  #pragma unroll
  for (int n = 0; n < 2; ++n)
    #pragma unroll
    for (int j = 0; j < 8; ++j) c[n][j] = 0.f;

  for (int t = 0; t < Sd; ++t) {
    const int s = dir ? (Sd - 1 - t) : t;
    {
      const u32x4* src = (const u32x4*)(xbf + (size_t)s * Bt * Ed);
      ((u32x4*)sh_x)[tid] = src[tid];
    }
    if (t + 1 < Sd) {
      const int sn = dir ? (Sd - 2 - t) : (t + 1);
      __builtin_prefetch(xbf + (size_t)sn * Bt * Ed + tid * 8, 0, 3);
    }
    __syncthreads();
    asm volatile("" ::: "memory");   // keep weight loads in-loop (no LICM/spill)

    v8f acc[4][2];
    #pragma unroll
    for (int g = 0; g < 4; ++g)
      #pragma unroll
      for (int n = 0; n < 2; ++n)
        acc[g][n] = (v8f){0.f,0.f,0.f,0.f,0.f,0.f,0.f,0.f};

    #pragma unroll
    for (int kc = 0; kc < 4; ++kc) {         // x part, K = 128
      FragBF a;
      a.u[0] = px[kc * 4];
      a.u[1] = px[kc * 4 + 2];
      #pragma unroll
      for (int g = 0; g < 4; ++g) {
        #pragma unroll
        for (int n = 0; n < 2; ++n) {
          FragBF bfr;
          bfr.u[0] = pih[g][n][kc * 4];
          bfr.u[1] = pih[g][n][kc * 4 + 2];
          acc[g][n] = __builtin_amdgcn_wmma_f32_16x16x32_bf16(
              false, a.v, false, bfr.v, (short)0, acc[g][n], false, false);
        }
      }
    }
    #pragma unroll
    for (int kc = 0; kc < 8; ++kc) {         // h part, K = 256
      FragBF a;
      a.u[0] = ph[kc * 4];
      a.u[1] = ph[kc * 4 + 2];
      #pragma unroll
      for (int g = 0; g < 4; ++g) {
        #pragma unroll
        for (int n = 0; n < 2; ++n) {
          FragBF bfr;
          bfr.u[0] = phh[g][n][kc * 4];
          bfr.u[1] = phh[g][n][kc * 4 + 2];
          acc[g][n] = __builtin_amdgcn_wmma_f32_16x16x32_bf16(
              false, a.v, false, bfr.v, (short)0, acc[g][n], false, false);
        }
      }
    }
    __syncthreads();

    const size_t so = (size_t)s * Bt * Hd;
    #pragma unroll
    for (int n = 0; n < 2; ++n) {
      __bf16* shh_o = n ? shh_o1 : shh_o0;
      __bf16* go_   = n ? gout1  : gout0;
      #pragma unroll
      for (int j = 0; j < 8; ++j) {
        const float gi = sigf (acc[0][n][j] + bgate[0][n]);
        const float gf = sigf (acc[1][n][j] + bgate[1][n]);
        const float gg = tanhf(acc[2][n][j] + bgate[2][n]);
        const float go = sigf (acc[3][n][j] + bgate[3][n]);
        const float cv = gf * c[n][j] + gi * gg;
        c[n][j] = cv;
        const float hv = go * tanhf(cv);
        shh_o[j * Hd]    = (__bf16)hv;
        go_[so + j * Hd] = (__bf16)hv;
      }
    }
  }
}

// ---------------- kernel 4: emissions feats = [hf|hb] @ W_out^T + b_out ----------------
__global__ __launch_bounds__(32)
void feats_kernel(const __bf16* __restrict__ hf, const __bf16* __restrict__ hb,
                  const __bf16* __restrict__ woutp, const float* __restrict__ b_out,
                  float* __restrict__ feats) {
  const int lane = threadIdx.x & 31;
  const int m  = lane & 15;
  const int kh = lane >> 4;
  const int r0 = blockIdx.x * 16;
  v8f acc = (v8f){0.f,0.f,0.f,0.f,0.f,0.f,0.f,0.f};

  const u32x4* pa_f = (const u32x4*)(hf + (size_t)(r0 + m) * Hd + kh * 8);
  const u32x4* pa_b = (const u32x4*)(hb + (size_t)(r0 + m) * Hd + kh * 8);
  const u32x4* pw   = (const u32x4*)(woutp + m * 512 + kh * 8);

  #pragma unroll
  for (int kc = 0; kc < 8; ++kc) {
    FragBF a, b;
    a.u[0] = pa_f[kc * 4];
    a.u[1] = pa_f[kc * 4 + 2];
    b.u[0] = pw[kc * 4];
    b.u[1] = pw[kc * 4 + 2];
    acc = __builtin_amdgcn_wmma_f32_16x16x32_bf16(
        false, a.v, false, b.v, (short)0, acc, false, false);
  }
  #pragma unroll
  for (int kc = 0; kc < 8; ++kc) {
    FragBF a, b;
    a.u[0] = pa_b[kc * 4];
    a.u[1] = pa_b[kc * 4 + 2];
    b.u[0] = pw[32 + kc * 4];
    b.u[1] = pw[32 + kc * 4 + 2];
    acc = __builtin_amdgcn_wmma_f32_16x16x32_bf16(
        false, a.v, false, b.v, (short)0, acc, false, false);
  }

  if (m < Td) {
    const float bo = b_out[m];
    #pragma unroll
    for (int j = 0; j < 8; ++j) {
      const int row = r0 + kh * 8 + j;
      feats[(size_t)row * Td + m] = acc[j] + bo;
    }
  }
}

// ---------------- kernel 5: CRF gold score + forward logZ ----------------
__global__ __launch_bounds__(576)
void crf_kernel(const int* __restrict__ sentence, const int* __restrict__ tags,
                const float* __restrict__ feats,
                const float* __restrict__ start_trans, const float* __restrict__ end_trans,
                const float* __restrict__ trans, float* __restrict__ out) {
  __shared__ float alpha[Bt][Td];
  __shared__ float numsh[Bt];
  __shared__ int   lensh[Bt];
  __shared__ float red[Bt];

  const int tid = threadIdx.x;     // 576 = 64 * 9
  const int b = tid / Td;
  const int j = tid % Td;

  alpha[b][j] = start_trans[j] + feats[b * Td + j];
  if (j == 0) {
    const int t0 = tags[b];
    numsh[b] = start_trans[t0] + feats[b * Td + t0];
    lensh[b] = 1;
  }
  __syncthreads();

  for (int s = 1; s < Sd; ++s) {
    const bool msk = sentence[s * Bt + b] != 0;
    float av[Td], mx = -1e30f;
    #pragma unroll
    for (int i = 0; i < Td; ++i) {
      av[i] = alpha[b][i] + trans[i * Td + j];
      mx = fmaxf(mx, av[i]);
    }
    float sum = 0.f;
    #pragma unroll
    for (int i = 0; i < Td; ++i) sum += __expf(av[i] - mx);
    const float nxt = mx + __logf(sum) + feats[((size_t)s * Bt + b) * Td + j];
    __syncthreads();
    if (msk) alpha[b][j] = nxt;
    if (j == 0 && msk) {
      const int pt = tags[(s - 1) * Bt + b];
      const int ct = tags[s * Bt + b];
      numsh[b] += trans[pt * Td + ct] + feats[((size_t)s * Bt + b) * Td + ct];
      lensh[b] += 1;
    }
    __syncthreads();
  }

  if (j == 0) {
    const int last = tags[(lensh[b] - 1) * Bt + b];
    const float num = numsh[b] + end_trans[last];
    float av[Td], mx = -1e30f;
    #pragma unroll
    for (int i = 0; i < Td; ++i) {
      av[i] = alpha[b][i] + end_trans[i];
      mx = fmaxf(mx, av[i]);
    }
    float sum = 0.f;
    #pragma unroll
    for (int i = 0; i < Td; ++i) sum += __expf(av[i] - mx);
    red[b] = -(num - (mx + __logf(sum)));
  }
  __syncthreads();
  if (tid == 0) {
    float acc = 0.f;
    for (int i = 0; i < Bt; ++i) acc += red[i];
    out[0] = acc;
  }
}

// ---------------- launcher ----------------
extern "C" void kernel_launch(void* const* d_in, const int* in_sizes, int n_in,
                              void* d_out, int out_size, void* d_ws, size_t ws_size,
                              hipStream_t stream) {
  const int*   sentence    = (const int*)d_in[0];
  const int*   tags        = (const int*)d_in[1];
  /* d_in[2] mask: unused — recomputed as sentence != PAD */
  const float* emb         = (const float*)d_in[3];
  const float* W_ih_f      = (const float*)d_in[4];
  const float* W_hh_f      = (const float*)d_in[5];
  const float* b_f         = (const float*)d_in[6];
  const float* W_ih_b      = (const float*)d_in[7];
  const float* W_hh_b      = (const float*)d_in[8];
  const float* b_b         = (const float*)d_in[9];
  const float* W_out       = (const float*)d_in[10];
  const float* b_out       = (const float*)d_in[11];
  const float* start_trans = (const float*)d_in[12];
  const float* end_trans   = (const float*)d_in[13];
  const float* trans       = (const float*)d_in[14];

  char* ws = (char*)d_ws;
  __bf16* xbf   = (__bf16*)(ws + 0);          // S*B*E bf16      = 8 MB
  __bf16* wihf  = (__bf16*)(ws + 8388608);    // 4H*E bf16       = 256 KB
  __bf16* whhf  = (__bf16*)(ws + 8650752);    // 4H*H bf16       = 512 KB
  __bf16* wihb  = (__bf16*)(ws + 9175040);
  __bf16* whhb  = (__bf16*)(ws + 9437184);
  __bf16* woutp = (__bf16*)(ws + 9961472);    // 16*512 bf16     = 16 KB
  __bf16* hfb   = (__bf16*)(ws + 9977856);    // S*B*H bf16      = 16 MB
  __bf16* hbb   = (__bf16*)(ws + 26755072);   // S*B*H bf16      = 16 MB
  float*  feats = (float*)(ws + 43532288);    // S*B*T f32       = 1.1 MB
  __bf16* gatf  = (__bf16*)(ws + 50331648);   // x-gates fwd     = 64 MB
  __bf16* gatb  = (__bf16*)(ws + 117440512);  // x-gates bwd     = 64 MB
  const size_t NEED_SPLIT = 184549376ull;     // 176 MB

  embed_kernel  <<<16384, 256, 0, stream>>>(sentence, emb, xbf);
  f2bf_kernel   <<<  512, 256, 0, stream>>>(W_ih_f, wihf, G4 * Ed);
  f2bf_kernel   <<< 1024, 256, 0, stream>>>(W_hh_f, whhf, G4 * Hd);
  f2bf_kernel   <<<  512, 256, 0, stream>>>(W_ih_b, wihb, G4 * Ed);
  f2bf_kernel   <<< 1024, 256, 0, stream>>>(W_hh_b, whhb, G4 * Hd);
  woutpad_kernel<<<   32, 256, 0, stream>>>(W_out, woutp);

  if (ws_size >= NEED_SPLIT) {
    // full-chip input GEMM + shortened serial recurrence
    xgates_kernel<<<dim3(2048, 2), 256, 0, stream>>>(xbf, wihf, wihb, gatf, gatb);
    bilstm_split_kernel<<<2, 1024, 0, stream>>>(gatf, gatb, whhf, b_f, whhb, b_b, hfb, hbb);
  } else {
    bilstm_kernel<<<2, 1024, 0, stream>>>(xbf, wihf, whhf, b_f, wihb, whhb, b_b, hfb, hbb);
  }
  feats_kernel <<<2048, 32, 0, stream>>>(hfb, hbb, woutp, b_out, feats);
  crf_kernel   <<<1, 576, 0, stream>>>(sentence, tags, feats, start_trans, end_trans, trans,
                                       (float*)d_out);
}
